// ProcessAudio_41111426957504
// MI455X (gfx1250) — matrix operand order repaired and checked by reading, not compiled, not run on hardware
//
#include <hip/hip_runtime.h>
#include <math.h>

// ---------------- problem constants ----------------
#define NS        4194304          // samples
#define NFFT      512
#define HOP       128
#define NFRAMES   32769            // 1 + NS/HOP
#define NBINS     257              // NFFT/2 + 1
#define MTILES    2049             // ceil(32769/16) -> padded M = 32784
#define FTILES    17               // ceil(257/16)   -> padded N = 272
#define KTILES    16               // 512 / 32
#define FRAG_H    512              // halves per fragment (32 lanes x 16)
#define A_TOTAL   ((size_t)MTILES * KTILES * FRAG_H)   // 16,785,408 halves
#define B_TOTAL   ((size_t)FTILES * KTILES * FRAG_H)   // 139,264 halves

typedef __attribute__((ext_vector_type(16))) _Float16 v16h;
typedef __attribute__((ext_vector_type(8)))  float    v8f;

// ---------------- workspace layout (bytes) ----------------
#define WS_PARTIALS  0                         // 1024 x float2 = 8192 B
#define WS_CSCALAR   8192                      // 1 float
#define WS_A         16384                     // A_TOTAL * 2 B
#define WS_BC        (16384 + 2*A_TOTAL)       // B_TOTAL * 2 B
#define WS_BS        (WS_BC + 2*B_TOTAL)       // B_TOTAL * 2 B

// ============ stage 1: deterministic mean / sumsq reduction ============
__global__ void k_reduce1(const float* __restrict__ x, float2* __restrict__ partials) {
    __shared__ float ssum[256], ssq[256];
    int tid = threadIdx.x;
    size_t stride = (size_t)gridDim.x * blockDim.x;
    float s = 0.f, q = 0.f;
    for (size_t i = (size_t)blockIdx.x * blockDim.x + tid; i < (size_t)NS; i += stride) {
        float v = x[i];
        s += v; q += v * v;
    }
    ssum[tid] = s; ssq[tid] = q;
    __syncthreads();
    for (int off = 128; off > 0; off >>= 1) {
        if (tid < off) { ssum[tid] += ssum[tid + off]; ssq[tid] += ssq[tid + off]; }
        __syncthreads();
    }
    if (tid == 0) partials[blockIdx.x] = make_float2(ssum[0], ssq[0]);
}

__global__ void k_reduce2(const float2* __restrict__ partials, float* __restrict__ cbuf) {
    __shared__ float ssum[256], ssq[256];
    int tid = threadIdx.x;
    float s = 0.f, q = 0.f;
    for (int k = 0; k < 4; ++k) {
        float2 p = partials[tid + 256 * k];
        s += p.x; q += p.y;
    }
    ssum[tid] = s; ssq[tid] = q;
    __syncthreads();
    for (int off = 128; off > 0; off >>= 1) {
        if (tid < off) { ssum[tid] += ssum[tid + off]; ssq[tid] += ssq[tid + off]; }
        __syncthreads();
    }
    if (tid == 0) {
        float sum = ssum[0], sumsq = ssq[0];
        float mean = sum / (float)NS;
        float var  = (sumsq - sum * mean) / (float)(NS - 1);   // ddof=1
        cbuf[0] = mean / sqrtf(var);   // reference bug: audio - (mean/std)
    }
}

// ============ stage 2: windowed frames -> A, pre-swizzled f16 fragments ============
// Fragment layout (16-bit A 16x32): lane L holds row M = L%16;
// half i -> K = (L<16 ? 0 : 8) + (i<8 ? i : i+8).
__global__ void k_prep(const float* __restrict__ audio, const float* __restrict__ cbuf,
                       _Float16* __restrict__ A) {
    size_t o = (size_t)blockIdx.x * blockDim.x + threadIdx.x;
    if (o >= A_TOTAL) return;
    int i    = (int)(o & 15);
    int l    = (int)((o >> 4) & 31);
    size_t frag = o >> 9;
    int kt   = (int)(frag & 15);
    int mt   = (int)(frag >> 4);
    int t    = mt * 16 + (l & 15);
    int kb   = ((l < 16) ? 0 : 8) + ((i < 8) ? i : i + 8);
    int n    = kt * 32 + kb;
    float val = 0.f;
    if (t < NFRAMES) {
        long p = (long)t * HOP + n - (NFFT / 2);     // reflect pad
        if (p < 0)  p = -p;
        if (p >= (long)NS) p = 2L * NS - 2 - p;
        float w = 0.5f * (1.0f - __cosf(6.28318530718f * (float)n / (float)NFFT));
        val = w * (audio[p] - cbuf[0]);
    }
    A[o] = (_Float16)val;
}

// ============ stage 3: DFT basis -> Bcos / Bsin, pre-swizzled f16 fragments ============
// Fragment layout (16-bit B 32x16): lane L holds col N = L%16;
// half i -> K = (L<16 ? 0 : 8) + (i<8 ? i : i+8).
__global__ void k_dft(_Float16* __restrict__ Bc, _Float16* __restrict__ Bs) {
    size_t o = (size_t)blockIdx.x * blockDim.x + threadIdx.x;
    if (o >= B_TOTAL) return;
    int i    = (int)(o & 15);
    int l    = (int)((o >> 4) & 31);
    int frag = (int)(o >> 9);
    int kt   = frag & 15;
    int ft   = frag >> 4;
    int f    = ft * 16 + (l & 15);
    int n    = kt * 32 + ((l < 16) ? 0 : 8) + ((i < 8) ? i : i + 8);
    float c = 0.f, s = 0.f;
    if (f < NBINS) {
        int m = (f * n) & (NFFT - 1);                 // exact modular reduction
        float ang = (float)m * (6.28318530718f / (float)NFFT);
        c = __cosf(ang);
        s = -__sinf(ang);                             // e^{-j 2 pi f n / N}
    }
    Bc[o] = (_Float16)c;
    Bs[o] = (_Float16)s;
}

// ============ stage 4: WMMA GEMM + fused epilogue ============
// One wave -> one 16(T) x 16(F) tile; two f32 accumulators (Re, Im).
__global__ void __launch_bounds__(256) k_gemm(const _Float16* __restrict__ A,
                                              const _Float16* __restrict__ Bc,
                                              const _Float16* __restrict__ Bs,
                                              float* __restrict__ out) {
    int wave = threadIdx.x >> 5;
    int lane = threadIdx.x & 31;
    int mt   = blockIdx.x * 8 + wave;
    if (mt >= MTILES) mt = MTILES - 1;   // clamp: duplicate tile, identical stores; EXEC stays all-ones
    int ft   = blockIdx.y;

    const _Float16* ap = A  + (((size_t)mt * KTILES) * 32 + lane) * 16;
    const _Float16* cp = Bc + (((size_t)ft * KTILES) * 32 + lane) * 16;
    const _Float16* sp = Bs + (((size_t)ft * KTILES) * 32 + lane) * 16;

    v8f acc_re = {};
    v8f acc_im = {};
    #pragma unroll
    for (int kt = 0; kt < KTILES; ++kt) {
        v16h a  = *(const v16h*)(ap);
        v16h bc = *(const v16h*)(cp);
        v16h bs = *(const v16h*)(sp);
        if (kt + 1 < KTILES)
            __builtin_prefetch(ap + FRAG_H, 0, 0);    // global_prefetch_b8 on A stream
        acc_re = __builtin_amdgcn_wmma_f32_16x16x32_f16(
            false, a, false, bc, (short)0, acc_re, false, false);
        acc_im = __builtin_amdgcn_wmma_f32_16x16x32_f16(
            false, a, false, bs, (short)0, acc_im, false, false);
        ap += FRAG_H; cp += FRAG_H; sp += FRAG_H;
    }

    // C/D layout: VGPR j -> M = j (lanes 0-15) / j+8 (lanes 16-31); N = lane%16
    int f = ft * 16 + (lane & 15);
    int row_off = (lane >= 16) ? 8 : 0;
    if (f < NBINS) {
        #pragma unroll
        for (int j = 0; j < 8; ++j) {
            int t = mt * 16 + j + row_off;
            if (t < NFRAMES) {
                float re = acc_re[j];
                float im = acc_im[j];
                float mag = sqrtf(re * re + im * im);
                float inv = 1.0f / fmaxf(mag, 1e-30f);
                size_t base = ((size_t)t * 3) * NBINS + f;
                out[base]             = __logf(mag + 1e-9f);  // log|X|
                out[base + NBINS]     = im * inv;             // sin(phase) == sin(unwrap(phase))
                out[base + 2 * NBINS] = re * inv;             // cos(phase) == cos(unwrap(phase))
            }
        }
    }
}

// ============ launcher ============
extern "C" void kernel_launch(void* const* d_in, const int* in_sizes, int n_in,
                              void* d_out, int out_size, void* d_ws, size_t ws_size,
                              hipStream_t stream) {
    (void)in_sizes; (void)n_in; (void)out_size; (void)ws_size;
    const float* audio = (const float*)d_in[0];
    float* out = (float*)d_out;
    char* ws = (char*)d_ws;

    float2*   partials = (float2*)  (ws + WS_PARTIALS);
    float*    cbuf     = (float*)   (ws + WS_CSCALAR);
    _Float16* Amat     = (_Float16*)(ws + WS_A);
    _Float16* Bcos     = (_Float16*)(ws + WS_BC);
    _Float16* Bsin     = (_Float16*)(ws + WS_BS);

    k_reduce1<<<1024, 256, 0, stream>>>(audio, partials);
    k_reduce2<<<1,    256, 0, stream>>>(partials, cbuf);

    k_prep<<<(int)((A_TOTAL + 255) / 256), 256, 0, stream>>>(audio, cbuf, Amat);
    k_dft <<<(int)((B_TOTAL + 255) / 256), 256, 0, stream>>>(Bcos, Bsin);

    dim3 grid((MTILES + 7) / 8, FTILES);
    k_gemm<<<grid, 256, 0, stream>>>(Amat, Bcos, Bsin, out);
}
